// ST2Block_70239895159312
// MI455X (gfx1250) — matrix operand order repaired and checked by reading, not compiled, not run on hardware
//
#include <hip/hip_runtime.h>
#include <hip/hip_bf16.h>
#include <cstdint>

// STGCN ST-block for MI455X (gfx1250, wave32, WMMA bf16 w/ f32 accum).
// Kernel 1: h[b,o,t,n] = sum_k Theta[k]^T @ (x[b,:,t,:] @ Lk[k]^T)   (fused)
// Kernel 2: temporal conv (as GEMM over (ic,kt)) + GLU gating.

typedef __attribute__((ext_vector_type(16))) __bf16        v16bf;
typedef __attribute__((ext_vector_type(8)))  float         v8f;
typedef __attribute__((ext_vector_type(4)))  unsigned int  u32x4;
typedef __attribute__((ext_vector_type(4)))  float         f32x4;

#define Bc  32
#define Cc  64
#define Tc  12
#define Nc  1024
#define Kc  3
#define KTc 3
#define Tout (Tc - KTc + 1)   // 10

// Native f32->bf16 (RNE); backend emits v_cvt_pk_bf16_f32-class ops.
__device__ __forceinline__ unsigned int pk_bf(float a, float b) {
  union { __bf16 h[2]; unsigned int u; } v;
  v.h[0] = (__bf16)a;
  v.h[1] = (__bf16)b;
  return v.u;
}
__device__ __forceinline__ unsigned short bf1(float a) {
  union { __bf16 h; unsigned short u; } v;
  v.h = (__bf16)a;
  return v.u;
}

// Build a 16-element bf16 fragment from two 16B LDS reads (ds_read_b128 x2).
__device__ __forceinline__ v16bf frag16(const unsigned short* p0,
                                        const unsigned short* p1) {
  union { u32x4 q[2]; v16bf v; } f;
  f.q[0] = *(const u32x4*)p0;
  f.q[1] = *(const u32x4*)p1;
  return f.v;
}

#define WMMA_BF16(a, b, c) \
  __builtin_amdgcn_wmma_f32_16x16x32_bf16(false, (a), false, (b), (short)0, (c), false, false)

// Pin "nds DS-reads then nw WMMAs" so B-fragments get distinct registers and
// the waits become partial dscnt values instead of full drains per WMMA.
#define SCHED_DS_THEN_WMMA(nds, nw)                      \
  do {                                                   \
    __builtin_amdgcn_sched_group_barrier(0x100, (nds), 0); \
    __builtin_amdgcn_sched_group_barrier(0x008, (nw), 0);  \
  } while (0)

// ---------------------------------------------------------------------------
// Kernel 1: spatial graph conv + channel mix.
// grid = (N/128, B*T), block = 256 (8 waves). LDS rows padded to 72 ushorts
// (144B, multiple of 16B -> aligned b128 fragment reads).
// Global->LDS staging is software-pipelined through registers.
// ---------------------------------------------------------------------------
__global__ __launch_bounds__(256) void stgcn_spatial(
    const float* __restrict__ x, const float* __restrict__ Lk,
    const float* __restrict__ Theta, float* __restrict__ h)
{
  __shared__ unsigned short Xs[64][72];    // x chunk   [c][m]   (9 KB)
  __shared__ unsigned short Ls[128][72];   // Lk chunk  [n][m]   (18 KB)
  __shared__ unsigned short Ths[64][72];   // Theta_k^T [o][c]   (9 KB)
  __shared__ unsigned short Ys[128][72];   // Y_k^T     [n][c]   (18 KB)

  const int tid  = threadIdx.x;
  const int lane = tid & 31;
  const int wid  = tid >> 5;
  const int lm   = lane & 15;
  const int hi8  = (lane >> 4) * 8;
  const int hi16 = (lane >> 4) * 16;

  const int b = blockIdx.y / Tc, t = blockIdx.y % Tc;
  const int n_base = blockIdx.x * 128;

  const int c0 = (wid & 3) * 16;    // stage-1 c-tile; stage-2 o-tile
  const int nh = (wid >> 2) * 64;   // n half within the 128-wide slice

  // Fixed per-thread staging slots (4 float4 for X, 8 float4 for Lk).
  int xc[4], xm[4], ln[8], lmm[8];
  #pragma unroll
  for (int j = 0; j < 4; ++j) {
    const int e4 = tid + j * 256;
    xc[j] = e4 >> 4; xm[j] = (e4 & 15) << 2;
  }
  #pragma unroll
  for (int j = 0; j < 8; ++j) {
    const int e4 = tid + j * 256;
    ln[j] = e4 >> 4; lmm[j] = (e4 & 15) << 2;
  }

  v8f Hacc[4] = { {}, {}, {}, {} };

  const float* xb = x + ((size_t)(b * Cc) * Tc + t) * Nc;

  for (int k = 0; k < Kc; ++k) {
    __syncthreads();                        // prev-k stage-2 done reading LDS
    // Theta_k^T: Ths[o][c] = Theta[k][c][o]  (float4 along o, scatter to LDS)
    const float* Th = Theta + (size_t)k * Cc * Cc;
    #pragma unroll
    for (int e4 = tid; e4 < 1024; e4 += 256) {       // 4 iterations
      const int c = e4 >> 4, o = (e4 & 15) << 2;
      f32x4 v = *(const f32x4*)(Th + c * Cc + o);
      Ths[o][c]     = bf1(v.x);
      Ths[o + 1][c] = bf1(v.y);
      Ths[o + 2][c] = bf1(v.z);
      Ths[o + 3][c] = bf1(v.w);
    }

    v8f Yacc[4] = { {}, {}, {}, {} };
    const float* Lkk = Lk + (size_t)k * Nc * Nc;

    // Pipeline prologue: fetch chunk mc=0 into registers.
    f32x4 xr[4], lr[8];
    #pragma unroll
    for (int j = 0; j < 4; ++j)
      xr[j] = *(const f32x4*)(xb + (size_t)xc[j] * Tc * Nc + xm[j]);
    #pragma unroll
    for (int j = 0; j < 8; ++j)
      lr[j] = *(const f32x4*)(Lkk + (size_t)(n_base + ln[j]) * Nc + lmm[j]);

    for (int mc = 0; mc < Nc; mc += 64) {
      __syncthreads();                      // all waves done reading prev chunk
      // Convert pipelined registers -> LDS (packed bf16 dword stores).
      #pragma unroll
      for (int j = 0; j < 4; ++j) {
        *(unsigned int*)&Xs[xc[j]][xm[j]]     = pk_bf(xr[j].x, xr[j].y);
        *(unsigned int*)&Xs[xc[j]][xm[j] + 2] = pk_bf(xr[j].z, xr[j].w);
      }
      #pragma unroll
      for (int j = 0; j < 8; ++j) {
        *(unsigned int*)&Ls[ln[j]][lmm[j]]     = pk_bf(lr[j].x, lr[j].y);
        *(unsigned int*)&Ls[ln[j]][lmm[j] + 2] = pk_bf(lr[j].z, lr[j].w);
      }
      __syncthreads();
      // Prefetch next chunk into registers; overlaps with WMMAs below.
      if (mc + 64 < Nc) {
        #pragma unroll
        for (int j = 0; j < 4; ++j)
          xr[j] = *(const f32x4*)(xb + (size_t)xc[j] * Tc * Nc + mc + 64 + xm[j]);
        #pragma unroll
        for (int j = 0; j < 8; ++j)
          lr[j] = *(const f32x4*)(Lkk + (size_t)(n_base + ln[j]) * Nc + mc + 64 + lmm[j]);
      }
      // Stage 1: Y[c][n] += X[c][m] * Lk[n][m].
      #pragma unroll
      for (int ks = 0; ks < 64; ks += 32) {
        v16bf a = frag16(&Xs[c0 + lm][ks + hi8], &Xs[c0 + lm][ks + 16 + hi8]);
        v16bf bf[4];
        #pragma unroll
        for (int nt = 0; nt < 4; ++nt) {
          const int nrow = nh + nt * 16 + lm;
          bf[nt] = frag16(&Ls[nrow][ks + hi8], &Ls[nrow][ks + 16 + hi8]);
        }
        #pragma unroll
        for (int nt = 0; nt < 4; ++nt)
          Yacc[nt] = WMMA_BF16(a, bf[nt], Yacc[nt]);
        SCHED_DS_THEN_WMMA(10, 4);
      }
    }

    __syncthreads();
    // Write Y transposed to LDS: Ys[n][c], packed bf16 (one b128 store each).
    #pragma unroll
    for (int nt = 0; nt < 4; ++nt) {
      const int nrow = nh + nt * 16 + lm;
      u32x4 pk;
      pk.x = pk_bf(Yacc[nt][0], Yacc[nt][1]);
      pk.y = pk_bf(Yacc[nt][2], Yacc[nt][3]);
      pk.z = pk_bf(Yacc[nt][4], Yacc[nt][5]);
      pk.w = pk_bf(Yacc[nt][6], Yacc[nt][7]);
      *(u32x4*)&Ys[nrow][c0 + hi8] = pk;
    }
    __syncthreads();
    // Stage 2: H[o][n] += Theta_k[c][o] * Y[c][n]   (K = 64 -> 2 steps)
    #pragma unroll
    for (int cs = 0; cs < 64; cs += 32) {
      v16bf a2 = frag16(&Ths[c0 + lm][cs + hi8], &Ths[c0 + lm][cs + 16 + hi8]);
      v16bf b2[4];
      #pragma unroll
      for (int nt = 0; nt < 4; ++nt) {
        const int nrow = nh + nt * 16 + lm;
        b2[nt] = frag16(&Ys[nrow][cs + hi8], &Ys[nrow][cs + 16 + hi8]);
      }
      #pragma unroll
      for (int nt = 0; nt < 4; ++nt)
        Hacc[nt] = WMMA_BF16(a2, b2[nt], Hacc[nt]);
      SCHED_DS_THEN_WMMA(10, 4);
    }
  }

  // Store h[b][o][t][n] (f32), per ISA C-layout: VGPR i -> M = i + 8*(lane>=16).
  #pragma unroll
  for (int nt = 0; nt < 4; ++nt) {
    const int n = n_base + nh + nt * 16 + lm;
    #pragma unroll
    for (int i = 0; i < 8; ++i) {
      const int o = c0 + hi8 + i;
      h[(((size_t)b * Cc + o) * Tc + t) * Nc + n] = Hacc[nt][i];
    }
  }
}

// ---------------------------------------------------------------------------
// Kernel 2: temporal conv as GEMM over (ic, kt) + in-register GLU gating.
// grid = (N/128, B*Tout), block = 256 (8 waves).
// ---------------------------------------------------------------------------
__global__ __launch_bounds__(256) void stgcn_temporal(
    const float* __restrict__ h, const float* __restrict__ w_t,
    const float* __restrict__ b_t, float* __restrict__ out)
{
  __shared__ unsigned short Ws[128][72];   // weights [oc][ic] for this kt (18 KB)
  __shared__ unsigned short Hs[128][72];   // h slice [n][ic] for this kt  (18 KB)

  const int tid  = threadIdx.x;
  const int lane = tid & 31;
  const int wid  = tid >> 5;
  const int lm   = lane & 15;
  const int hi8  = (lane >> 4) * 8;
  const int hi16 = (lane >> 4) * 16;

  const int b = blockIdx.y / Tout, tp = blockIdx.y % Tout;
  const int n_base = blockIdx.x * 128;
  const int o0 = (wid & 3) * 16;
  const int nh = (wid >> 2) * 64;

  v8f pacc[4] = { {}, {}, {}, {} };   // oc in [0,64)   (p half)
  v8f qacc[4] = { {}, {}, {}, {} };   // oc in [64,128) (q half)

  for (int kt = 0; kt < KTc; ++kt) {
    __syncthreads();
    // Ws[oc][ic] = w_t[oc][ic][kt]   (w_t strides: oc:192, ic:3, kt:1)
    for (int e = tid; e < 128 * 64; e += 256) {
      const int oc = e >> 6, ic = e & 63;
      Ws[oc][ic] = bf1(w_t[(size_t)(oc * 64 + ic) * KTc + kt]);
    }
    // Hs[nn][ic] = h[b][ic][tp+kt][n_base+nn]  (float4 along n, scatter to LDS)
    #pragma unroll
    for (int e4 = tid; e4 < 2048; e4 += 256) {       // 8 iterations
      const int ic = e4 >> 5, nn = (e4 & 31) << 2;
      f32x4 v = *(const f32x4*)(h + (((size_t)b * Cc + ic) * Tc + tp + kt) * Nc
                                   + n_base + nn);
      Hs[nn][ic]     = bf1(v.x);
      Hs[nn + 1][ic] = bf1(v.y);
      Hs[nn + 2][ic] = bf1(v.z);
      Hs[nn + 3][ic] = bf1(v.w);
    }
    __syncthreads();
    #pragma unroll
    for (int cs = 0; cs < 64; cs += 32) {
      v16bf ap = frag16(&Ws[o0 + lm][cs + hi8],      &Ws[o0 + lm][cs + 16 + hi8]);
      v16bf aq = frag16(&Ws[64 + o0 + lm][cs + hi8], &Ws[64 + o0 + lm][cs + 16 + hi8]);
      v16bf bf[4];
      #pragma unroll
      for (int nt = 0; nt < 4; ++nt) {
        const int nrow = nh + nt * 16 + lm;
        bf[nt] = frag16(&Hs[nrow][cs + hi8], &Hs[nrow][cs + 16 + hi8]);
      }
      #pragma unroll
      for (int nt = 0; nt < 4; ++nt) {
        pacc[nt] = WMMA_BF16(ap, bf[nt], pacc[nt]);
        qacc[nt] = WMMA_BF16(aq, bf[nt], qacc[nt]);
      }
      SCHED_DS_THEN_WMMA(12, 8);
    }
  }

  // GLU: out = (p + b_p + x_in) * sigmoid(q + b_q), x_in = h[b,o,tp+2,n] (f32)
  #pragma unroll
  for (int nt = 0; nt < 4; ++nt) {
    const int n = n_base + nh + nt * 16 + lm;
    #pragma unroll
    for (int i = 0; i < 8; ++i) {
      const int op = o0 + hi8 + i;
      const float xin = h[(((size_t)b * Cc + op) * Tc + tp + KTc - 1) * Nc + n];
      const float p = pacc[nt][i] + b_t[op] + xin;
      const float q = qacc[nt][i] + b_t[op + 64];
      out[(((size_t)b * Cc + op) * Tout + tp) * Nc + n] =
          p * (1.0f / (1.0f + __expf(-q)));
    }
  }
}

extern "C" void kernel_launch(void* const* d_in, const int* in_sizes, int n_in,
                              void* d_out, int out_size, void* d_ws, size_t ws_size,
                              hipStream_t stream) {
  (void)in_sizes; (void)n_in; (void)out_size; (void)ws_size;
  const float* x     = (const float*)d_in[0];   // (B,C,T,N)
  const float* Lk    = (const float*)d_in[1];   // (K,N,N)
  const float* Theta = (const float*)d_in[2];   // (K,C,C)
  const float* w_t   = (const float*)d_in[3];   // (2C,C,KT,1)
  const float* b_t   = (const float*)d_in[4];   // (2C,)
  float* out = (float*)d_out;                   // (B,C,Tout,N) f32
  float* h   = (float*)d_ws;                    // scratch: B*C*T*N f32 = 96 MiB

  dim3 g1(Nc / 128, Bc * Tc);
  stgcn_spatial<<<g1, 256, 0, stream>>>(x, Lk, Theta, h);
  dim3 g2(Nc / 128, Bc * Tout);
  stgcn_temporal<<<g2, 256, 0, stream>>>(h, w_t, b_t, out);
}
